// RNNModel_42030549959157
// MI455X (gfx1250) — matrix operand order repaired and checked by reading, not compiled, not run on hardware
//
#include <hip/hip_runtime.h>
#include <hip/hip_bf16.h>

typedef __attribute__((ext_vector_type(16))) _Float16 v16h;
typedef __attribute__((ext_vector_type(8)))  float    v8f;
typedef __attribute__((ext_vector_type(2)))  _Float16 h2;

#define B_ 256
#define T_ 1024
#define I_ 2
#define H_ 128
#define L_ 4
#define LDH (H_ + 8)   // padded LDS row stride in halves (272B rows -> bank spread)

union Frag16 { v16h v; uint4 u[2]; };
union Frag8f { v8f v; uint4 u[2]; float f[8]; };
union Pack8h { _Float16 h[8]; h2 p[4]; uint4 u; };

// ---------------------------------------------------------------------------
// Convert weights fp32 -> fp16 once.
__global__ void cvt_weights(const float* __restrict__ Whh, const float* __restrict__ Wih,
                            _Float16* __restrict__ WhhH, _Float16* __restrict__ WihH) {
    int idx = blockIdx.x * 256 + threadIdx.x;
    if (idx < L_ * H_ * H_)        WhhH[idx] = (_Float16)Whh[idx];
    if (idx < (L_ - 1) * H_ * H_)  WihH[idx] = (_Float16)Wih[idx];
}

// ---------------------------------------------------------------------------
// Layer-0 input projection (I=2, scalar) written directly in the (transposed)
// WMMA C-fragment layout used by the recurrence:
//   index = (((t*16 + bt)*8 + nt)*32 + lane)*8 + r
//   value = xp[b = bt*16 + (lane&15)][n = nt*16 + (lane>>4)*8 + r]
__global__ void xp_layer0(const float* __restrict__ x, const float* __restrict__ Wih0,
                          const float* __restrict__ bih, const float* __restrict__ bhh,
                          float* __restrict__ xp) {
    long gid  = (long)blockIdx.x * 256 + threadIdx.x;
    int  r    = (int)(gid & 7);
    int  lane = (int)((gid >> 3) & 31);
    int  nt   = (int)((gid >> 8) & 7);
    long btt  = gid >> 11;            // t*16 + bt
    int  bt   = (int)(btt & 15);
    int  t    = (int)(btt >> 4);
    int  hi   = lane >> 4;
    int  b    = bt * 16 + (lane & 15);       // batch row (D layout: lane = N = batch)
    int  n    = nt * 16 + hi * 8 + r;        // feature   (D layout: VGPR r = M = n)
    const float* xv = x + ((long)b * T_ + t) * I_;
    float v = xv[0] * Wih0[n * I_ + 0] + xv[1] * Wih0[n * I_ + 1] + bih[n] + bhh[n];
    xp[gid] = v;
}

// ---------------------------------------------------------------------------
// Recurrence computed transposed:  h_t^T = relu(xp_t^T + Whh x h_{t-1}^T).
//   A = Whh tile (resident in registers for all 1024 steps, A-layout)
//   B = h^T     -> per-lane fragment is a CONTIGUOUS 32B row chunk of LDS h
//   D = h_new^T -> per-lane 8 relu'd halves are contiguous: 1 ds_store_b128
// One barrier per step; h(t) streamed out via GLOBAL_STORE_ASYNC_FROM_LDS
// (ASYNCcnt), drained two steps later before the ping-pong buffer is reused.
__global__ __launch_bounds__(256, 1)
void rnn_recurrence(const float* __restrict__ xp, const _Float16* __restrict__ Whh,
                    _Float16* __restrict__ hseq, int store_all) {
    __shared__ _Float16 hbuf[2][64 * LDH];
    const int tid   = threadIdx.x;
    const int lane  = tid & 31;
    const int w     = tid >> 5;        // wave id == n-tile
    const int lh    = lane & 15;
    const int hi    = lane >> 4;       // lane half
    const int bBase = blockIdx.x * 64;

    // Resident A fragments: rows n = w*16+lh of Whh.
    // A layout: lanes 0-15 K = ks*32+0..15; lanes 16-31 K = ks*32+16..31.
    Frag16 afr[4];
    {
        const _Float16* wrow = Whh + (w * 16 + lh) * H_ + hi * 16;
        #pragma unroll
        for (int ks = 0; ks < 4; ++ks) {
            afr[ks].u[0] = *(const uint4*)(wrow + ks * 32);
            afr[ks].u[1] = *(const uint4*)(wrow + ks * 32 + 8);
        }
    }

    for (int i = tid; i < 64 * LDH; i += 256) hbuf[0][i] = (_Float16)0.f;
    __syncthreads();

    const h2 zero2 = {(_Float16)0.f, (_Float16)0.f};
    int cur = 0;
    for (int t = 0; t < T_; ++t) {
        const _Float16* hc = hbuf[cur];
        _Float16*       hn = hbuf[cur ^ 1];

        // Drain this wave's async stores from step t-2 (they read the buffer
        // we are about to overwrite). Combined with the per-step barrier this
        // guarantees all waves' t-2 stores are done before any t write.
        asm volatile("s_wait_asynccnt 0x0" ::: "memory");

        if (t + 1 < T_) {  // prefetch next step's xp fragments
            const float* pf = xp + (((((long)(t + 1) * 16 + blockIdx.x * 4) * 8) + w) << 8) + lane * 8;
            __builtin_prefetch(pf, 0, 1);
        }

        #pragma unroll
        for (int bt = 0; bt < 4; ++bt) {
            // accumulator initialized from xp fragment (contiguous 32B/lane)
            Frag8f c;
            const float* cf = xp + (((((long)t * 16 + blockIdx.x * 4 + bt) * 8) + w) << 8) + lane * 8;
            c.u[0] = *(const uint4*)(cf);
            c.u[1] = *(const uint4*)(cf + 4);
            // B fragments = h_prev^T: lane has row b = bt*16+lh, contiguous k.
            // B layout: lanes 0-15 K = ks*32+0..15; lanes 16-31 K = ks*32+16..31.
            Frag16 b[4];
            const _Float16* bp = hc + (bt * 16 + lh) * LDH + hi * 16;
            #pragma unroll
            for (int ks = 0; ks < 4; ++ks) {
                b[ks].u[0] = *(const uint4*)(bp + ks * 32);
                b[ks].u[1] = *(const uint4*)(bp + ks * 32 + 8);
            }
            #pragma unroll
            for (int ks = 0; ks < 4; ++ks)
                c.v = __builtin_amdgcn_wmma_f32_16x16x32_f16(
                        false, afr[ks].v, false, b[ks].v, (short)0, c.v, false, false);
            // cvt (RNE) -> packed relu (v_pk_max) -> ONE contiguous b128 store:
            // lane's 8 values are h_new[b = bt*16+lh][n = w*16+hi*8 .. +7]
            Pack8h pk;
            #pragma unroll
            for (int j = 0; j < 8; ++j) pk.h[j] = (_Float16)c.f[j];
            #pragma unroll
            for (int j = 0; j < 4; ++j) pk.p[j] = __builtin_elementwise_max(pk.p[j], zero2);
            *(uint4*)(hn + (bt * 16 + lh) * LDH + w * 16 + hi * 8) = pk.u;
        }
        __syncthreads();   // h(t) complete in LDS and visible to all waves

        // stream h(t) -> hseq[t][bBase..+64][:] directly from LDS (ASYNCcnt)
        if (store_all || t == T_ - 1) {
            _Float16* dst = hseq + ((long)t * B_ + bBase) * H_;
            #pragma unroll
            for (int it = 0; it < 4; ++it) {
                int chunk = it * 256 + tid;        // 1024 chunks of 8 halves
                int row = chunk >> 4;
                int col = (chunk & 15) * 8;
                unsigned long long ga = (unsigned long long)(dst + row * H_ + col);
                unsigned           lo = (unsigned)(unsigned long long)(const void*)(hn + row * LDH + col);
                asm volatile("global_store_async_from_lds_b128 %0, %1, off"
                             :: "v"(ga), "v"(lo) : "memory");
            }
        }
        cur ^= 1;
    }
}

// ---------------------------------------------------------------------------
// Input projection for layers 1..3:  xp^T = Wih x hseq^T + b  (N=128 x M=262144).
// Wave nt keeps its Wih A-fragments resident and loops over 8 row-tiles;
// B streamed as contiguous row chunks of hseq; D stored in fragment layout.
__global__ __launch_bounds__(256)
void inproj_gemm(const _Float16* __restrict__ hseq, const _Float16* __restrict__ Wih,
                 const float* __restrict__ bih, const float* __restrict__ bhh,
                 float* __restrict__ xp) {
    const int lane = threadIdx.x & 31;
    const int nt   = threadIdx.x >> 5;
    const int lh   = lane & 15;
    const int hi   = lane >> 4;

    Frag16 afr[4];
    {
        const _Float16* wrow = Wih + (nt * 16 + lh) * H_ + hi * 16;
        #pragma unroll
        for (int ks = 0; ks < 4; ++ks) {
            afr[ks].u[0] = *(const uint4*)(wrow + ks * 32);
            afr[ks].u[1] = *(const uint4*)(wrow + ks * 32 + 8);
        }
    }
    // bias for this lane's 8 feature rows n = nt*16 + hi*8 + r
    float bias[8];
    #pragma unroll
    for (int r = 0; r < 8; ++r) {
        int n = nt * 16 + hi * 8 + r;
        bias[r] = bih[n] + bhh[n];
    }

    long mbase = (long)blockIdx.x * 8;
    #pragma unroll 2
    for (int i = 0; i < 8; ++i) {
        long mtile = mbase + i;           // 16 consecutive rows (same t)
        Frag8f c;
        #pragma unroll
        for (int r = 0; r < 8; ++r) c.f[r] = bias[r];
        const _Float16* brow = hseq + (mtile * 16 + lh) * H_ + hi * 16;
        #pragma unroll
        for (int ks = 0; ks < 4; ++ks) {
            Frag16 b;
            b.u[0] = *(const uint4*)(brow + ks * 32);
            b.u[1] = *(const uint4*)(brow + ks * 32 + 8);
            c.v = __builtin_amdgcn_wmma_f32_16x16x32_f16(
                    false, afr[ks].v, false, b.v, (short)0, c.v, false, false);
        }
        float* dst = xp + ((mtile * 8 + nt) << 8) + lane * 8;
        *(uint4*)(dst)     = c.u[0];
        *(uint4*)(dst + 4) = c.u[1];
    }
}

// ---------------------------------------------------------------------------
// Final FC on last timestep: out[b] = hseq[T-1][b][:] . fc_w + fc_b   (C=1)
__global__ void final_fc(const _Float16* __restrict__ hlast, const float* __restrict__ fcw,
                         const float* __restrict__ fcb, float* __restrict__ out) {
    int b = blockIdx.x * blockDim.x + threadIdx.x;
    if (b >= B_) return;
    const _Float16* h = hlast + (long)b * H_;
    float s = 0.f;
    #pragma unroll 8
    for (int k = 0; k < H_; ++k) s += (float)h[k] * fcw[k];
    out[b] = s + fcb[0];
}

// ---------------------------------------------------------------------------
extern "C" void kernel_launch(void* const* d_in, const int* in_sizes, int n_in,
                              void* d_out, int out_size, void* d_ws, size_t ws_size,
                              hipStream_t stream) {
    const float* x     = (const float*)d_in[0];
    const float* W_ih0 = (const float*)d_in[1];
    const float* W_ihl = (const float*)d_in[2];   // [L-1, H, H]
    const float* W_hh  = (const float*)d_in[3];   // [L, H, H]
    const float* b_ih  = (const float*)d_in[4];   // [L, H]
    const float* b_hh  = (const float*)d_in[5];   // [L, H]
    const float* fc_w  = (const float*)d_in[6];
    const float* fc_b  = (const float*)d_in[7];
    float* out = (float*)d_out;

    // workspace carve-up
    float*    xp    = (float*)d_ws;                                  // B*T*H f32 (frag layout)
    _Float16* hseq  = (_Float16*)(xp + (size_t)B_ * T_ * H_);        // [T][B][H] f16
    _Float16* WhhH  = hseq + (size_t)B_ * T_ * H_;                   // L*H*H f16
    _Float16* WihH  = WhhH + (size_t)L_ * H_ * H_;                   // (L-1)*H*H f16

    cvt_weights<<<(L_ * H_ * H_ + 255) / 256, 256, 0, stream>>>(W_hh, W_ihl, WhhH, WihH);

    xp_layer0<<<(B_ * T_ * H_) / 256, 256, 0, stream>>>(x, W_ih0, b_ih, b_hh, xp);

    for (int l = 0; l < L_; ++l) {
        int store_all = (l + 1 < L_) ? 1 : 0;   // last layer only needs t=T-1
        rnn_recurrence<<<B_ / 64, 256, 0, stream>>>(
            xp, WhhH + (size_t)l * H_ * H_, hseq, store_all);
        if (l + 1 < L_) {
            inproj_gemm<<<(T_ * B_ / 16) / 8, 256, 0, stream>>>(
                hseq, WihH + (size_t)l * H_ * H_,
                b_ih + (size_t)(l + 1) * H_, b_hh + (size_t)(l + 1) * H_, xp);
        }
    }

    final_fc<<<1, 256, 0, stream>>>(hseq + (size_t)(T_ - 1) * B_ * H_, fc_w, fc_b, out);
}
